// DeepFactor_33767032881269
// MI455X (gfx1250) — matrix operand order repaired and checked by reading, not compiled, not run on hardware
//
#include <hip/hip_runtime.h>
#include <hip/hip_bf16.h>

// ---------------------------------------------------------------------------
// DeepFactor: K=10 relu-LSTM branches (U=64) over B=32, T=1024, D=32,
// shared Dense(1) head, mean over K.  MI455X (gfx1250, wave32) design:
//   scan kernel: grid (K, 2 batch-halves), 256 thr = 8 wave32 / WG.
//   Per step: z[16,256] = [h_t | x_t](16x96) x [U_rec ; W](96x256) + b
//   via 3 chained v_wmma_f32_16x16x32_f16 per 16x16 tile (2 tiles/wave).
//   Weights preloaded to registers (f16); h/z staged through LDS; c in regs.
//   Gate sigmoids use hardware v_tanh_f32 (or v_rcp_f32 fallback) to keep
//   the serial per-step latency chain short.
// ---------------------------------------------------------------------------

namespace {
constexpr int Bk = 32;    // batch
constexpr int Tk = 1024;  // timesteps
constexpr int Dk = 32;    // input dim
constexpr int Uk = 64;    // hidden units
constexpr int Kk = 10;    // factor branches
constexpr int COLS = 4 * Uk;  // 256 gate columns (i|f|c|o)
}

typedef __attribute__((ext_vector_type(16))) _Float16 v16h;
typedef __attribute__((ext_vector_type(8)))  _Float16 v8h;
typedef __attribute__((ext_vector_type(8)))  float    v8f;
typedef __attribute__((ext_vector_type(4)))  float    v4f;

__device__ __forceinline__ float sigmoidf_fast(float v) {
#if __has_builtin(__builtin_amdgcn_tanhf)
    // sigmoid(x) = 0.5*tanh(x/2) + 0.5 : v_mul + v_tanh_f32 + v_fma
    return fmaf(0.5f, __builtin_amdgcn_tanhf(0.5f * v), 0.5f);
#else
    // single v_rcp_f32 instead of the IEEE divide fix-up sequence
    return __builtin_amdgcn_rcpf(1.0f + __expf(-v));
#endif
}

__launch_bounds__(256)
__global__ void deepfactor_scan_kernel(const float* __restrict__ x,
                                       const float* __restrict__ W,
                                       const float* __restrict__ Urec,
                                       const float* __restrict__ bias,
                                       const float* __restrict__ Wd,
                                       float* __restrict__ ypart)
{
    const int k    = blockIdx.x;   // factor branch 0..9
    const int half = blockIdx.y;   // batch half 0..1 (16 rows each)
    const int tid  = threadIdx.x;
    const int wave = tid >> 5;     // 0..7
    const int lane = tid & 31;
    const int hi   = lane >> 4;    // lane half
    const int l15  = lane & 15;

    __shared__ _Float16 hbuf[16 * Uk];    // hidden state h_t   (f16)  2 KB
    __shared__ float    zbuf[16 * COLS];  // gate preactivations      16 KB
    __shared__ float    ybuf[8][16];      // per-wave head partials

    // ---- one-time preload: WMMA B operands (f16) for this wave's 2 tiles ---
    // B layout (32x16, KxN): elems 0..7 -> K = hi*8+e,
    //                        elems 8..15 -> K = 16+hi*8+(e-8).
    v16h Bop[2][3];     // [col tile j][K-chunk q: u0..31, u32..63, d0..31]
    float biasv[2];
    #pragma unroll
    for (int j = 0; j < 2; ++j) {
        const int col = 32 * wave + 16 * j + l15;
        biasv[j] = bias[k * COLS + col];
        #pragma unroll
        for (int q = 0; q < 3; ++q) {
            #pragma unroll
            for (int e = 0; e < 16; ++e) {
                const int row = (e < 8) ? (hi * 8 + e) : (16 + hi * 8 + (e - 8));
                const float v = (q < 2)
                    ? Urec[(k * Uk + 32 * q + row) * COLS + col]
                    : W[(k * Dk + row) * COLS + col];
                Bop[j][q][e] = (_Float16)v;
            }
        }
    }

    // ---- elementwise-phase assignment: thread -> (batch row, 4 units) -----
    const int eb = tid & 15;   // batch row 0..15
    const int eg = tid >> 4;   // unit group 0..15
    const int u0 = eg * 4;
    float wdv[4];
    #pragma unroll
    for (int i = 0; i < 4; ++i) wdv[i] = Wd[u0 + i];
    float creg[4] = {0.f, 0.f, 0.f, 0.f};   // cell state lives in registers

    // zero initial hidden state
    for (int i = tid; i < 16 * Uk; i += 256) hbuf[i] = (_Float16)0.0f;
    __syncthreads();

    const int bglob = half * 16 + l15;            // global batch row for A
    const float* xrow = x + (size_t)bglob * Tk * Dk;
    const int d0 = hi * 8;                        // A-layout K sub-offset

    for (int t = 0; t < Tk; ++t) {
        // prefetch next step's x slice toward the WGP caches
        __builtin_prefetch(&xrow[(t + 1) * Dk + d0], 0, 3);

        // A operand from x_t (input-projection K-chunk), f32 -> f16.
        // A layout (16-bit 16x32): elems 0..7 -> K = d0..d0+7,
        //                          elems 8..15 -> K = 16+d0 .. 16+d0+7.
        v16h Ax;
        #pragma unroll
        for (int e = 0; e < 8; ++e) Ax[e]     = (_Float16)xrow[t * Dk + d0 + e];
        #pragma unroll
        for (int e = 0; e < 8; ++e) Ax[8 + e] = (_Float16)xrow[t * Dk + 16 + d0 + e];

        // A operands from hidden state in LDS (two 16-byte ds loads each)
        v16h Ah[2];
        #pragma unroll
        for (int q = 0; q < 2; ++q) {
            const v8h a0 = *(const v8h*)&hbuf[l15 * Uk + 32 * q + hi * 8];
            const v8h a1 = *(const v8h*)&hbuf[l15 * Uk + 32 * q + hi * 8 + 16];
            #pragma unroll
            for (int e = 0; e < 8; ++e) { Ah[q][e] = a0[e]; Ah[q][8 + e] = a1[e]; }
        }

        // z = b + x_t*W + h*U_rec : 3-deep WMMA chain per column tile
        #pragma unroll
        for (int j = 0; j < 2; ++j) {
            v8f acc;
            #pragma unroll
            for (int v = 0; v < 8; ++v) acc[v] = biasv[j];
            acc = __builtin_amdgcn_wmma_f32_16x16x32_f16(false, Ax,    false, Bop[j][2], (short)0, acc, false, false);
            acc = __builtin_amdgcn_wmma_f32_16x16x32_f16(false, Ah[0], false, Bop[j][0], (short)0, acc, false, false);
            acc = __builtin_amdgcn_wmma_f32_16x16x32_f16(false, Ah[1], false, Bop[j][1], (short)0, acc, false, false);
            // C/D layout: elem v -> row (hi*8 + v), col = lane&15
            const int col = 32 * wave + 16 * j + l15;
            #pragma unroll
            for (int v = 0; v < 8; ++v) zbuf[(hi * 8 + v) * COLS + col] = acc[v];
        }
        __syncthreads();   // z visible; h reads of this step complete

        // ---- LSTM cell update (relu candidate / relu cell output) ---------
        const v4f zi = *(const v4f*)&zbuf[eb * COLS + u0];
        const v4f zf = *(const v4f*)&zbuf[eb * COLS + Uk + u0];
        const v4f zc = *(const v4f*)&zbuf[eb * COLS + 2 * Uk + u0];
        const v4f zo = *(const v4f*)&zbuf[eb * COLS + 3 * Uk + u0];
        float s = 0.f;
        #pragma unroll
        for (int i = 0; i < 4; ++i) {
            const float ig   = sigmoidf_fast(zi[i]);
            const float fg   = sigmoidf_fast(zf[i]);
            const float cand = fmaxf(zc[i], 0.f);
            creg[i] = fg * creg[i] + ig * cand;
            const float hval = sigmoidf_fast(zo[i]) * fmaxf(creg[i], 0.f);
            hbuf[eb * Uk + u0 + i] = (_Float16)hval;
            s += hval * wdv[i];
        }

        // ---- Dense(1) head partial: reduce 16 unit-groups per batch row ---
        s += __shfl_down(s, 16, 32);        // pair groups within the wave
        if (lane < 16) ybuf[wave][lane] = s;
        __syncthreads();   // h_t+1 + ybuf visible
        if (wave == 0 && lane < 16) {
            float acc = 0.f;
            #pragma unroll
            for (int w2 = 0; w2 < 8; ++w2) acc += ybuf[w2][lane];
            ypart[((size_t)(k * 2 + half) * Tk + t) * 16 + lane] = acc;
        }
    }
}

// out[b,t] = mean_k( head partials ) + bd
__global__ void deepfactor_head_kernel(const float* __restrict__ ypart,
                                       const float* __restrict__ bd,
                                       float* __restrict__ out)
{
    const int idx = blockIdx.x * blockDim.x + threadIdx.x;
    if (idx >= Bk * Tk) return;
    const int b = idx / Tk;
    const int t = idx - b * Tk;
    const int half = b >> 4;
    const int b16  = b & 15;
    float s = 0.f;
    #pragma unroll
    for (int k = 0; k < Kk; ++k)
        s += ypart[((size_t)(k * 2 + half) * Tk + t) * 16 + b16];
    out[idx] = s * (1.0f / Kk) + bd[0];
}

extern "C" void kernel_launch(void* const* d_in, const int* in_sizes, int n_in,
                              void* d_out, int out_size, void* d_ws, size_t ws_size,
                              hipStream_t stream) {
    const float* x    = (const float*)d_in[0];   // [B, T, D]
    const float* W    = (const float*)d_in[1];   // [K, D, 4U]
    const float* Urec = (const float*)d_in[2];   // [K, U, 4U]
    const float* bias = (const float*)d_in[3];   // [K, 4U]
    const float* Wd   = (const float*)d_in[4];   // [U, 1]
    const float* bd   = (const float*)d_in[5];   // [1]
    float* out   = (float*)d_out;                // [B, T, 1]
    float* ypart = (float*)d_ws;                 // [K*2, T, 16] = 1.25 MiB

    dim3 grid(Kk, 2, 1);
    deepfactor_scan_kernel<<<grid, 256, 0, stream>>>(x, W, Urec, bias, Wd, ypart);

    const int n = Bk * Tk;
    deepfactor_head_kernel<<<(n + 255) / 256, 256, 0, stream>>>(ypart, bd, out);
}
